// AASIST_61168924230019
// MI455X (gfx1250) — matrix operand order repaired and checked by reading, not compiled, not run on hardware
//
#include <hip/hip_runtime.h>
#include <hip/hip_bf16.h>
#include <cstdint>
#include <cstddef>

// ---------------------------------------------------------------------------
// CDNA5 (gfx1250) AASIST pipeline: every conv is a bf16 WMMA GEMM
// (v_wmma_f32_16x16x32_bf16). Intermediates live in bf16 (model is HBM-bound:
// ~0.7GB @ 23.3TB/s). Encoder convs stage their input tiles in LDS via
// GLOBAL_LOAD_ASYNC_TO_LDS_B32 (ASYNCcnt) and feed WMMA from ds loads.
// ---------------------------------------------------------------------------

typedef __bf16 bf16_t;
typedef __attribute__((ext_vector_type(16))) __bf16 v16bf;
typedef __attribute__((ext_vector_type(8)))  float  v8f;

__device__ __forceinline__ float selu_f(float x) {
  const float a = 1.6732632423543772f, s = 1.0507009873554805f;
  return x > 0.f ? s * x : s * a * (expf(x) - 1.f);
}

__device__ __forceinline__ v8f wmma_bf16(v16bf a, v16bf b, v8f c) {
  // 8 args: (neg_a, A, neg_b, B, c_mod, C, reuse_a, reuse_b)
  return __builtin_amdgcn_wmma_f32_16x16x32_bf16(false, a, false, b,
                                                 (short)0, c, false, false);
}

// A fragment (16x32 bf16, M x K), contiguous K order (matches B loaders):
// lanes 0-15: M=lane, K = {0..15}; lanes 16-31: M=lane-16, K={16..31}
__device__ __forceinline__ v16bf load_a_frag(const bf16_t* __restrict__ A,
                                             int lda, int m0, int k0, int lane) {
  int m = m0 + (lane & 15);
  int half = (lane < 16) ? 0 : 16;
  const bf16_t* p = A + (size_t)m * lda + k0 + half;
  v16bf a;
#pragma unroll
  for (int i = 0; i < 16; ++i) a[i] = p[i];
  return a;
}

// B fragment (32x16 bf16, K x N) from an LDS-staged 1D signal window:
// B[k][n] = xs[n + k]  (sinc conv im2col, Cin = 1)
__device__ __forceinline__ v16bf load_b_frag_lds(const bf16_t* xs, int k0, int lane) {
  int n  = lane & 15;
  int kb = k0 + ((lane < 16) ? 0 : 16);
  v16bf b;
#pragma unroll
  for (int i = 0; i < 16; ++i) b[i] = xs[n + kb + i];
  return b;
}

// B fragment from the LDS conv tile: tile[r][j] holds in[r][t0 + j - 2],
// row stride 132. k = ci*KW + dk;  t = t0 + nl + dk - KW/2.
template <int KW>
__device__ __forceinline__ v16bf load_b_frag_tile(const bf16_t* tile, int nl0,
                                                  int k0, int lane) {
  int nl = nl0 + (lane & 15);
  int kb = k0 + ((lane < 16) ? 0 : 16);
  v16bf b;
#pragma unroll
  for (int i = 0; i < 16; ++i) {
    int k  = kb + i;
    int ci = k / KW;
    int dk = k - ci * KW;
    b[i] = tile[ci * 132 + nl + dk + (KW == 3 ? 1 : 2)];
  }
  return b;
}

// ---------------------------------------------------------------------------
// Device-side sinc filter bank -> bf16 A matrix [80][160] (M pad 70->80, K pad
// 129->160 so the WMMA K-loop is 5 clean chunks of 32).
// ---------------------------------------------------------------------------
__global__ void build_sinc_kernel(const float* __restrict__ low_hz,
                                  const float* __restrict__ band_hz,
                                  bf16_t* __restrict__ A) {
  const int Kpad = 160, CoutPad = 80;
  int idx = blockIdx.x * blockDim.x + threadIdx.x;
  if (idx >= CoutPad * Kpad) return;
  int k = idx % Kpad, c = idx / Kpad;
  float val = 0.f;
  if (c < 70 && k < 129) {
    float low  = 50.f + fabsf(low_hz[c]);
    float high = fminf(fmaxf(low + 50.f + fabsf(band_hz[c]), 50.f), 8000.f);
    float band = high - low;
    if (k == 64) {
      val = 1.f;  // 2*band / (2*band)
    } else {
      int   j   = (k < 64) ? k : (128 - k);
      float n   = 2.f * 3.14159265358979f * (float)(j - 64) / 16000.f;
      float lin = 63.5f * (float)j / 63.f;
      float win = 0.54f - 0.46f * cosf(2.f * 3.14159265358979f * lin / 129.f);
      val = (sinf(high * n) - sinf(low * n)) / (n * 0.5f) * win / (2.f * band);
    }
  }
  A[idx] = (bf16_t)val;
}

// Pack fp32 conv weights [Cout][Cin][kw] -> bf16 [Cout][Kpad] (zero K padding)
__global__ void pack_weights_kernel(const float* __restrict__ w,
                                    bf16_t* __restrict__ A, int Kin, int Kpad,
                                    int total) {
  int idx = blockIdx.x * blockDim.x + threadIdx.x;
  if (idx >= total) return;
  int k = idx % Kpad, c = idx / Kpad;
  A[idx] = (bf16_t)((k < Kin) ? w[c * Kin + k] : 0.f);
}

// Fuse conv-bias + eval BatchNorm into per-channel scale/shift.
__global__ void fuse_bn_kernel(const float* __restrict__ bias,
                               const float* __restrict__ g,
                               const float* __restrict__ bb,
                               const float* __restrict__ m,
                               const float* __restrict__ v,
                               float* __restrict__ scale,
                               float* __restrict__ shift, int C) {
  int c = threadIdx.x;
  if (c < C) {
    float s  = g[c] * rsqrtf(v[c] + 1e-5f);
    float b0 = bias ? bias[c] : 0.f;
    scale[c] = s;
    shift[c] = (b0 - m[c]) * s + bb[c];
  }
}

// ---------------------------------------------------------------------------
// Sinc conv (Cin=1, K=129, pad 64) as WMMA: 5 waves/block = 5 channel tiles
// (70 channels padded to 80), one 16-sample output tile per block, input
// window staged once in LDS as bf16. Epilogue: BN + SELU + |.| (feeds maxpool)
// ---------------------------------------------------------------------------
__global__ void sinc_conv_kernel(const float* __restrict__ x,
                                 const bf16_t* __restrict__ A,
                                 const float* __restrict__ scale,
                                 const float* __restrict__ shift,
                                 bf16_t* __restrict__ out) {
  const int L = 64512, PAD = 64, Kpad = 160, Cout = 70;
  __shared__ bf16_t xs[16 + 160];  // window [t0-64, t0+15+95]
  int b  = blockIdx.y;
  int t0 = blockIdx.x * 16;
  const float* xb = x + (size_t)b * L;
  for (int i = threadIdx.x; i < 176; i += blockDim.x) {
    int   t = t0 - PAD + i;
    float v = (t >= 0 && t < L) ? xb[t] : 0.f;
    xs[i] = (bf16_t)v;
  }
  __syncthreads();

  int lane = threadIdx.x & 31;
  int mt   = threadIdx.x >> 5;  // 0..4 channel tile
  v8f acc = {};
#pragma unroll
  for (int kc = 0; kc < Kpad; kc += 32) {
    v16bf af = load_a_frag(A, Kpad, mt * 16, kc, lane);
    v16bf bf = load_b_frag_lds(xs, kc, lane);
    acc = wmma_bf16(af, bf, acc);
  }
  int n     = t0 + (lane & 15);
  int mbase = mt * 16 + ((lane < 16) ? 0 : 8);
#pragma unroll
  for (int i = 0; i < 8; ++i) {
    int m = mbase + i;
    if (m < Cout) {
      float y = acc[i] * scale[m] + shift[m];
      y = fabsf(selu_f(y));
      out[((size_t)b * Cout + m) * L + n] = (bf16_t)y;
    }
  }
}

// ---------------------------------------------------------------------------
// Generic conv1d-as-WMMA (kw=3 pad1 or kw=1). 8 waves/block, block owns
// (one Cout-tile) x (128 output positions). Input tile [rows][132] staged in
// LDS: interior via GLOBAL_LOAD_ASYNC_TO_LDS_B32 (ASYNCcnt), halos/pad rows
// scalar. Epilogue: scale/shift [+res] [+selu] -> bf16.
// ---------------------------------------------------------------------------
template <int KW, bool DO_SELU, bool DO_RES>
__global__ void conv_wmma_kernel(const bf16_t* __restrict__ in,
                                 const bf16_t* __restrict__ A,
                                 const float* __restrict__ scale,
                                 const float* __restrict__ shift,
                                 const bf16_t* __restrict__ res,
                                 bf16_t* __restrict__ out, int Cin, int Cout,
                                 int L, int Kpad) {
  extern __shared__ bf16_t tile[];       // [rows][132]; col j <-> t = t0 + j - 2
  const int rows = (Kpad - 1) / KW + 1;  // padded channel rows (>= Cin)
  int lane = threadIdx.x & 31;
  int wid  = threadIdx.x >> 5;
  int t0   = blockIdx.x * 128;
  int mt   = blockIdx.y;
  int b    = blockIdx.z;
  const bf16_t* inb = in + (size_t)b * Cin * L;

  bool interior = (t0 >= 1) && (t0 + 128 < L);  // uniform per block
  if (interior) {
    // Aligned interior [t0, t0+128) of every real channel row: async VMEM->LDS.
    unsigned long long gbase = (unsigned long long)(uintptr_t)(inb + t0);
    unsigned int lbase = (unsigned int)(uintptr_t)tile;
    for (int idx = threadIdx.x; idx < Cin * 64; idx += 256) {
      int r = idx >> 6, c2 = idx & 63;
      unsigned int ldsb = lbase + (unsigned)r * 264u + 4u + 4u * (unsigned)c2;
      unsigned int voff = (unsigned)r * (unsigned)L * 2u + 4u * (unsigned)c2;
      asm volatile("global_load_async_to_lds_b32 %0, %1, %2"
                   :: "v"(ldsb), "v"(voff), "s"(gbase)
                   : "memory");
    }
    // Halo columns j=1 (t0-1) and j=130 (t0+128).
    for (int idx = threadIdx.x; idx < Cin * 2; idx += 256) {
      int r = idx >> 1;
      int j = (idx & 1) ? 130 : 1;
      tile[r * 132 + j] = inb[(size_t)r * L + t0 + j - 2];
    }
    // Zero the K-padding rows.
    for (int idx = threadIdx.x; idx < (rows - Cin) * 130; idx += 256) {
      int r = Cin + idx / 130, j = idx % 130 + 1;
      tile[r * 132 + j] = (bf16_t)0.f;
    }
    asm volatile("s_wait_asynccnt 0x0" ::: "memory");
  } else {
    for (int idx = threadIdx.x; idx < rows * 130; idx += 256) {
      int r = idx / 130, j = idx % 130 + 1;
      int t = t0 + j - 2;
      bf16_t v = (bf16_t)0.f;
      if (r < Cin && t >= 0 && t < L) v = inb[(size_t)r * L + t];
      tile[r * 132 + j] = v;
    }
  }
  __syncthreads();

  int nl0 = wid * 16;  // this wave's 16-sample slice of the 128-wide tile
  v8f acc = {};
  for (int kc = 0; kc < Kpad; kc += 32) {
    v16bf af = load_a_frag(A, Kpad, mt * 16, kc, lane);
    v16bf bf = load_b_frag_tile<KW>(tile, nl0, kc, lane);
    acc = wmma_bf16(af, bf, acc);
  }
  int n     = t0 + nl0 + (lane & 15);
  int mbase = mt * 16 + ((lane < 16) ? 0 : 8);
#pragma unroll
  for (int i = 0; i < 8; ++i) {
    int   m = mbase + i;
    float y = acc[i] * scale[m] + shift[m];
    if (DO_RES) y += (float)res[((size_t)b * Cout + m) * L + n];
    if (DO_SELU) y = selu_f(y);
    out[((size_t)b * Cout + m) * L + n] = (bf16_t)y;
  }
}

// maxpool1d(3,3) over already-|.| data
__global__ void maxpool3_kernel(const bf16_t* __restrict__ in,
                                bf16_t* __restrict__ out, int Lin, int Lout,
                                size_t total) {
  size_t idx = (size_t)blockIdx.x * blockDim.x + threadIdx.x;
  if (idx >= total) return;
  size_t t = idx % Lout, bc = idx / Lout;
  const bf16_t* p = in + bc * (size_t)Lin + t * 3;
  float a = (float)p[0], b = (float)p[1], c = (float)p[2];
  out[idx] = (bf16_t)fmaxf(a, fmaxf(b, c));
}

// exact adaptive_avg_pool1d(L/2) for even L
__global__ void avgpool2_kernel(const bf16_t* __restrict__ in,
                                bf16_t* __restrict__ out, int Lout,
                                size_t total) {
  size_t idx = (size_t)blockIdx.x * blockDim.x + threadIdx.x;
  if (idx >= total) return;
  size_t t = idx % Lout, bc = idx / Lout;
  const bf16_t* p = in + bc * (size_t)(2 * Lout) + 2 * t;
  out[idx] = (bf16_t)(0.5f * ((float)p[0] + (float)p[1]));
}

// per-(batch,channel) mean over L -> nodes[b][node][c]
__global__ void chan_mean_kernel(const bf16_t* __restrict__ feat,
                                 float* __restrict__ nodes, int C, int L,
                                 int node) {
  int bc = blockIdx.x;
  int c = bc % C, b = bc / C;
  const bf16_t* p = feat + ((size_t)b * C + c) * L;
  float s = 0.f;
  for (int i = threadIdx.x; i < L; i += blockDim.x) s += (float)p[i];
  __shared__ float red[256];
  red[threadIdx.x] = s;
  __syncthreads();
  for (int w = 128; w > 0; w >>= 1) {
    if (threadIdx.x < w) red[threadIdx.x] += red[threadIdx.x + w];
    __syncthreads();
  }
  if (threadIdx.x == 0) nodes[((size_t)b * 2 + node) * C + c] = red[0] / (float)L;
}

// ---------------------------------------------------------------------------
// Tiny head: GAT(128->64) -> GAT(64->32) -> MLP(64->32->2), one block / batch
// ---------------------------------------------------------------------------
__global__ void head_kernel(const float* __restrict__ nodes, const float* g1w,
                            const float* g1b, const float* g1a,
                            const float* g1ab, const float* g2w,
                            const float* g2b, const float* g2a,
                            const float* g2ab, const float* cw1,
                            const float* cb1, const float* cw2,
                            const float* cb2, float* __restrict__ out) {
  __shared__ float h1[2][64], o1[2][64], h2[2][32], fl[64], hid[32];
  __shared__ float at1[2][2], at2[2][2];
  int b = blockIdx.x, t = threadIdx.x;
  const float* nb = nodes + (size_t)b * 256;

  for (int n = 0; n < 2; ++n) {
    float acc = g1b[t];
    for (int i = 0; i < 128; ++i) acc += nb[n * 128 + i] * g1w[i * 64 + t];
    h1[n][t] = acc;
  }
  __syncthreads();
  if (t == 0) {
    float si[2], sj[2];
    for (int n = 0; n < 2; ++n) {
      float a = 0.f, c = 0.f;
      for (int d = 0; d < 64; ++d) { a += h1[n][d] * g1a[d]; c += h1[n][d] * g1a[64 + d]; }
      si[n] = a; sj[n] = c;
    }
    for (int i2 = 0; i2 < 2; ++i2) {
      float s0 = si[i2] + sj[0] + g1ab[0], s1 = si[i2] + sj[1] + g1ab[0];
      s0 = ((s0 > 0.f) ? s0 : 0.2f * s0) * 0.5f;
      s1 = ((s1 > 0.f) ? s1 : 0.2f * s1) * 0.5f;
      float mx = fmaxf(s0, s1);
      float e0 = expf(s0 - mx), e1 = expf(s1 - mx), d = e0 + e1;
      at1[i2][0] = e0 / d; at1[i2][1] = e1 / d;
    }
  }
  __syncthreads();
  for (int n = 0; n < 2; ++n) {
    float v = at1[n][0] * h1[0][t] + at1[n][1] * h1[1][t];
    o1[n][t] = (v > 0.f) ? v : (expf(v) - 1.f);
  }
  __syncthreads();
  if (t < 32) {
    for (int n = 0; n < 2; ++n) {
      float acc = g2b[t];
      for (int i = 0; i < 64; ++i) acc += o1[n][i] * g2w[i * 32 + t];
      h2[n][t] = acc;
    }
  }
  __syncthreads();
  if (t == 0) {
    float si[2], sj[2];
    for (int n = 0; n < 2; ++n) {
      float a = 0.f, c = 0.f;
      for (int d = 0; d < 32; ++d) { a += h2[n][d] * g2a[d]; c += h2[n][d] * g2a[32 + d]; }
      si[n] = a; sj[n] = c;
    }
    for (int i2 = 0; i2 < 2; ++i2) {
      float s0 = si[i2] + sj[0] + g2ab[0], s1 = si[i2] + sj[1] + g2ab[0];
      s0 = ((s0 > 0.f) ? s0 : 0.2f * s0) * 0.5f;
      s1 = ((s1 > 0.f) ? s1 : 0.2f * s1) * 0.5f;
      float mx = fmaxf(s0, s1);
      float e0 = expf(s0 - mx), e1 = expf(s1 - mx), d = e0 + e1;
      at2[i2][0] = e0 / d; at2[i2][1] = e1 / d;
    }
  }
  __syncthreads();
  if (t < 32) {
    for (int n = 0; n < 2; ++n) {
      float v = at2[n][0] * h2[0][t] + at2[n][1] * h2[1][t];
      fl[n * 32 + t] = (v > 0.f) ? v : (expf(v) - 1.f);
    }
  }
  __syncthreads();
  if (t < 32) {
    float acc = cb1[t];
    for (int i = 0; i < 64; ++i) acc += fl[i] * cw1[i * 32 + t];
    hid[t] = selu_f(acc);
  }
  __syncthreads();
  if (t < 2) {
    float acc = cb2[t];
    for (int i = 0; i < 32; ++i) acc += hid[i] * cw2[i * 2 + t];
    out[b * 2 + t] = acc;
  }
}

// ---------------------------------------------------------------------------
// Host orchestration (graph-capture safe: all launches on `stream`,
// deterministic bump allocation inside d_ws, no hipMalloc/sync).
// ---------------------------------------------------------------------------
extern "C" void kernel_launch(void* const* d_in, const int* in_sizes, int n_in,
                              void* d_out, int out_size, void* d_ws,
                              size_t ws_size, hipStream_t stream) {
  (void)in_sizes; (void)n_in; (void)out_size; (void)ws_size;
  auto F = [&](int i) -> const float* { return (const float*)d_in[i]; };

  char* wsb = (char*)d_ws;
  size_t off = 0;
  auto alloc = [&](size_t bytes) -> void* {
    off = (off + 255) & ~(size_t)255;
    void* p = wsb + off;
    off += bytes;
    return p;
  };

  const int B = 16, Lx = 64512, Lp = 21504, L2 = 10752, L3 = 5376;

  bf16_t* A_sinc  = (bf16_t*)alloc((size_t)80 * 160 * 2);
  float*  sc_sinc = (float*)alloc(70 * 4);
  float*  sh_sinc = (float*)alloc(70 * 4);
  bf16_t* h_full  = (bf16_t*)alloc((size_t)B * 70 * Lx * 2);
  bf16_t* h_pool  = (bf16_t*)alloc((size_t)B * 70 * Lp * 2);
  const size_t BUFE = (size_t)B * 64 * Lp;  // covers all activation shapes
  bf16_t* buf0 = (bf16_t*)alloc(BUFE * 2);
  bf16_t* buf1 = (bf16_t*)alloc(BUFE * 2);
  bf16_t* buf2 = (bf16_t*)alloc(BUFE * 2);
  bf16_t* buf3 = (bf16_t*)alloc(BUFE * 2);
  float*  nodes = (float*)alloc((size_t)B * 2 * 128 * 4);

  bf16_t* A_ds1 = (bf16_t*)alloc((size_t)64 * 96 * 2);
  bf16_t* A_c11 = (bf16_t*)alloc((size_t)64 * 224 * 2);
  bf16_t* A_c21 = (bf16_t*)alloc((size_t)64 * 192 * 2);
  bf16_t* A_c12 = (bf16_t*)alloc((size_t)64 * 192 * 2);
  bf16_t* A_c22 = (bf16_t*)alloc((size_t)64 * 192 * 2);
  bf16_t* A_c13 = (bf16_t*)alloc((size_t)128 * 192 * 2);
  bf16_t* A_c23 = (bf16_t*)alloc((size_t)128 * 384 * 2);
  bf16_t* A_ds3 = (bf16_t*)alloc((size_t)128 * 64 * 2);
  float*  scbuf = (float*)alloc((size_t)8 * 256 * 4);
  auto SC = [&](int li) { return scbuf + li * 256; };
  auto SH = [&](int li) { return scbuf + li * 256 + 128; };

  auto pack = [&](int wi, int Cout, int Kin, int Kpad, bf16_t* A) {
    int total = Cout * Kpad;
    pack_weights_kernel<<<dim3((total + 255) / 256), dim3(256), 0, stream>>>(
        F(wi), A, Kin, Kpad, total);
  };
  auto fuse = [&](int bi, int bni, float* s, float* h, int C) {
    fuse_bn_kernel<<<dim3(1), dim3(128), 0, stream>>>(
        bi < 0 ? nullptr : F(bi), F(bni), F(bni + 1), F(bni + 2), F(bni + 3),
        s, h, C);
  };
  auto conv = [&](const bf16_t* in, const bf16_t* A, const float* s,
                  const float* h, const bf16_t* res, bf16_t* outp, int Cin,
                  int Cout, int L, int Kpad, int kw, bool dres) {
    int rows = (Kpad - 1) / kw + 1;
    size_t shmem = (size_t)rows * 132 * sizeof(bf16_t);
    dim3 grid(L / 128, Cout / 16, B), blk(256);
    if (kw == 1)
      conv_wmma_kernel<1, false, false><<<grid, blk, shmem, stream>>>(
          in, A, s, h, nullptr, outp, Cin, Cout, L, Kpad);
    else if (dres)
      conv_wmma_kernel<3, true, true><<<grid, blk, shmem, stream>>>(
          in, A, s, h, res, outp, Cin, Cout, L, Kpad);
    else
      conv_wmma_kernel<3, true, false><<<grid, blk, shmem, stream>>>(
          in, A, s, h, nullptr, outp, Cin, Cout, L, Kpad);
  };

  // ---- Sinc front end: build filters, WMMA conv, BN+SELU+|.|, maxpool3 ----
  build_sinc_kernel<<<dim3((80 * 160 + 255) / 256), dim3(256), 0, stream>>>(
      F(1), F(2), A_sinc);
  fuse(-1, 3, sc_sinc, sh_sinc, 70);
  sinc_conv_kernel<<<dim3(Lx / 16, B), dim3(160), 0, stream>>>(
      F(0), A_sinc, sc_sinc, sh_sinc, h_full);
  {
    size_t total = (size_t)B * 70 * Lp;
    maxpool3_kernel<<<dim3((unsigned)((total + 255) / 256)), dim3(256), 0,
                      stream>>>(h_full, h_pool, Lx, Lp, total);
  }

  // ---- Two encoders (spec @ param base 7, temp @ 55), feat-mean -> nodes ---
  for (int e = 0; e < 2; ++e) {
    int base = (e == 0) ? 7 : 55;
    pack(base + 12, 64, 70, 96, A_ds1);   fuse(base + 13, base + 14, SC(0), SH(0), 64);
    pack(base + 0, 64, 210, 224, A_c11);  fuse(base + 1, base + 2, SC(1), SH(1), 64);
    pack(base + 6, 64, 192, 192, A_c21);  fuse(base + 7, base + 8, SC(2), SH(2), 64);
    pack(base + 18, 64, 192, 192, A_c12); fuse(base + 19, base + 20, SC(3), SH(3), 64);
    pack(base + 24, 64, 192, 192, A_c22); fuse(base + 25, base + 26, SC(4), SH(4), 64);
    pack(base + 30, 128, 192, 192, A_c13); fuse(base + 31, base + 32, SC(5), SH(5), 128);
    pack(base + 36, 128, 384, 384, A_c23); fuse(base + 37, base + 38, SC(6), SH(6), 128);
    pack(base + 42, 128, 64, 64, A_ds3);   fuse(base + 43, base + 44, SC(7), SH(7), 128);

    // block1: 70 -> 64 @ L=21504 (downsample ident)
    conv(h_pool, A_ds1, SC(0), SH(0), nullptr, buf0, 70, 64, Lp, 96, 1, false);
    conv(h_pool, A_c11, SC(1), SH(1), nullptr, buf1, 70, 64, Lp, 224, 3, false);
    conv(buf1, A_c21, SC(2), SH(2), buf0, buf2, 64, 64, Lp, 192, 3, true);
    {
      size_t tot = (size_t)B * 64 * L2;
      avgpool2_kernel<<<dim3((unsigned)((tot + 255) / 256)), dim3(256), 0,
                        stream>>>(buf2, buf3, L2, tot);
    }
    // block2: 64 -> 64 @ L=10752 (identity ident)
    conv(buf3, A_c12, SC(3), SH(3), nullptr, buf0, 64, 64, L2, 192, 3, false);
    conv(buf0, A_c22, SC(4), SH(4), buf3, buf1, 64, 64, L2, 192, 3, true);
    {
      size_t tot = (size_t)B * 64 * L3;
      avgpool2_kernel<<<dim3((unsigned)((tot + 255) / 256)), dim3(256), 0,
                        stream>>>(buf1, buf2, L3, tot);
    }
    // block3: 64 -> 128 @ L=5376 (downsample ident); feat = block output
    conv(buf2, A_ds3, SC(7), SH(7), nullptr, buf0, 64, 128, L3, 64, 1, false);
    conv(buf2, A_c13, SC(5), SH(5), nullptr, buf1, 64, 128, L3, 192, 3, false);
    conv(buf1, A_c23, SC(6), SH(6), buf0, buf3, 128, 128, L3, 384, 3, true);

    chan_mean_kernel<<<dim3(B * 128), dim3(256), 0, stream>>>(buf3, nodes, 128,
                                                              L3, e);
  }

  // ---- GAT x2 + classifier head ----
  head_kernel<<<dim3(B), dim3(64), 0, stream>>>(
      nodes, F(103), F(104), F(105), F(106), F(107), F(108), F(109), F(110),
      F(111), F(112), F(113), F(114), (float*)d_out);
}
